// LC_82712480186994
// MI455X (gfx1250) — matrix operand order repaired and checked by reading, not compiled, not run on hardware
//
#include <hip/hip_runtime.h>
#include <stdint.h>

// ---------------------------------------------------------------------------
// MI455X (gfx1250) fused cosine-sim top-5 contrastive loss.
//
//   K1: row-L2-normalize z (fp32) -> x (f16, row-major [N][128])   ~4 MB in ws
//   K2: per 32-row strip per wave: s = x . x^T via v_wmma_f32_16x16x32_f16
//       (K=128 as 4 chained WMMAs per 16-row group, 8 per tile). B fragments
//       are double-buffered in registers; top-5 fast-reject uses __any()
//       (one v_cmp + scalar branch, no cross-lane traffic).
//   K3: per-row pos/neg logits + loss terms from fp32 z (wave per row)
//   K4: deterministic single-block reduction -> d_out[0]
// ---------------------------------------------------------------------------

typedef __attribute__((ext_vector_type(16))) _Float16 v16h;
typedef __attribute__((ext_vector_type(8)))  _Float16 v8h;
typedef __attribute__((ext_vector_type(4)))  _Float16 v4h;
typedef __attribute__((ext_vector_type(8)))  float    v8f;
typedef __attribute__((ext_vector_type(4)))  float    v4f;

#define N_ROWS 16384
#define DIMS   128
#define TOPK   5
#define NEG_INF (-3.4e38f)

// ---------------------------------------------------------------------------
// K1: x[row] = z[row] / max(||z[row]||2, 1e-12), stored as f16.
__global__ __launch_bounds__(256)
void lc_normalize(const float* __restrict__ z, _Float16* __restrict__ x) {
  const int wave = threadIdx.x >> 5;
  const int lane = threadIdx.x & 31;
  const int row  = blockIdx.x * 8 + wave;
  const float* zr = z + (size_t)row * DIMS + lane * 4;
  v4f v = *(const v4f*)zr;
  float ss = v[0]*v[0] + v[1]*v[1] + v[2]*v[2] + v[3]*v[3];
  #pragma unroll
  for (int off = 16; off > 0; off >>= 1) ss += __shfl_xor(ss, off);
  const float inv = 1.0f / fmaxf(sqrtf(ss), 1e-12f);
  v4h h;
  #pragma unroll
  for (int j = 0; j < 4; ++j) h[j] = (_Float16)(v[j] * inv);
  *(v4h*)(x + (size_t)row * DIMS + lane * 4) = h;
}

// ---------------------------------------------------------------------------
// K2: one wave per 32-row strip (two 16-row WMMA groups sharing B fragments).
__global__ __launch_bounds__(32)
void lc_sim_topk(const _Float16* __restrict__ x, int* __restrict__ top_idx) {
  __shared__ float tile[32][17];             // padded: conflict-free row reads

  const int lane    = threadIdx.x;           // 0..31
  const int r0      = blockIdx.x * 32;
  const int halfSel = lane >> 4;
  const int sub     = lane & 15;

  // A fragments (16-bit A 16x32): lane l -> row M = l&15;
  // lanes 0-15 hold K 0..15, lanes 16-31 hold K 16..31 (16 contiguous halfs).
  const int kbA = halfSel * 16;
  v16h a0[4], a1[4];
  #pragma unroll
  for (int f = 0; f < 4; ++f) {
    const _Float16* p0 = x + (size_t)(r0 + sub) * DIMS + f * 32 + kbA;
    const _Float16* p1 = p0 + 16 * DIMS;
    v8h lo0 = *(const v8h*)p0,        hi0 = *(const v8h*)(p0 + 8);
    v8h lo1 = *(const v8h*)p1,        hi1 = *(const v8h*)(p1 + 8);
    #pragma unroll
    for (int j = 0; j < 8; ++j) {
      a0[f][j] = lo0[j]; a0[f][j + 8] = hi0[j];
      a1[f][j] = lo1[j]; a1[f][j + 8] = hi1[j];
    }
  }

  // Per-row top-5 state: lane l owns row r0 + l.
  float t0 = NEG_INF, t1 = NEG_INF, t2 = NEG_INF, t3 = NEG_INF, t4 = NEG_INF;
  int   i0 = 0, i1 = 0, i2 = 0, i3 = 0, i4 = 0;
  float thresh = NEG_INF;                    // wave-uniform min over rows' t4

  // B fragments (16-bit B 32x16): lane l -> col N = l&15;
  // lanes 0-15 hold K 0..15, lanes 16-31 hold K 16..31 (16 contiguous halfs).
  const int kbB = halfSel * 16;

#define LOAD_B(dst, ccc) do {                                            \
    const _Float16* pb_ = x + (size_t)((ccc) + sub) * DIMS + kbB;        \
    (dst)[0] = *(const v16h*)(pb_);                                      \
    (dst)[1] = *(const v16h*)(pb_ + 32);                                 \
    (dst)[2] = *(const v16h*)(pb_ + 64);                                 \
    (dst)[3] = *(const v16h*)(pb_ + 96);                                 \
  } while (0)

  // One 16-column tile: 8 WMMAs, ballot fast-reject, rare LDS slow path.
  auto process = [&](int c0, const v16h* b) {
    v8f acc0 = {}, acc1 = {};
    #pragma unroll
    for (int f = 0; f < 4; ++f) {
      acc0 = __builtin_amdgcn_wmma_f32_16x16x32_f16(
          false, a0[f], false, b[f], (short)0, acc0, false, false);
      acc1 = __builtin_amdgcn_wmma_f32_16x16x32_f16(
          false, a1[f], false, b[f], (short)0, acc1, false, false);
    }
    // Per-lane max over this lane's 16 tile elements; __any -> v_cmp + branch.
    float m = fmaxf(acc0[0], acc1[0]);
    #pragma unroll
    for (int j = 1; j < 8; ++j) m = fmaxf(m, fmaxf(acc0[j], acc1[j]));
    if (__any(m > thresh)) {                 // wave-uniform scalar branch
      #pragma unroll
      for (int j = 0; j < 8; ++j) {
        tile[(halfSel << 3) + j][sub]      = acc0[j];
        tile[16 + (halfSel << 3) + j][sub] = acc1[j];
      }
      __syncthreads();
      #pragma unroll
      for (int c = 0; c < 16; ++c) {         // all 32 lanes: own row insertion
        float v = tile[lane][c];
        if (v > t4) {
          int g = c0 + c;
          if (v > t2) {
            if (v > t1) {
              if (v > t0) { t4=t3;i4=i3; t3=t2;i3=i2; t2=t1;i2=i1; t1=t0;i1=i0; t0=v;i0=g; }
              else        { t4=t3;i4=i3; t3=t2;i3=i2; t2=t1;i2=i1; t1=v;i1=g; }
            } else        { t4=t3;i4=i3; t3=t2;i3=i2; t2=v;i2=g; }
          } else {
            if (v > t3)   { t4=t3;i4=i3; t3=v;i3=g; }
            else          { t4=v;i4=g; }
          }
        }
      }
      __syncthreads();
      float t = t4;                          // min over all 32 rows
      #pragma unroll
      for (int off = 16; off > 0; off >>= 1) t = fminf(t, __shfl_xor(t, off));
      thresh = t;
    }
  };

  // Double-buffered sweep over 1024 column tiles, unrolled by two so the
  // buffers rotate without register copies.
  v16h bufA[4], bufB[4];
  LOAD_B(bufA, 0);
  for (int c0 = 0; c0 < N_ROWS; c0 += 32) {
    LOAD_B(bufB, c0 + 16);
    if (c0 + 32 < N_ROWS)
      __builtin_prefetch(x + (size_t)(c0 + 32 + sub) * DIMS + kbB, 0, 1);
    process(c0, bufA);
    if (c0 + 32 < N_ROWS) LOAD_B(bufA, c0 + 32);
    process(c0 + 16, bufB);
  }
#undef LOAD_B

  int* o = top_idx + (size_t)(r0 + lane) * TOPK;
  o[0] = i0; o[1] = i1; o[2] = i2; o[3] = i3; o[4] = i4;
}

// ---------------------------------------------------------------------------
// K3: per-row loss. One wave per row; each lane owns 4 contiguous dims.
__global__ __launch_bounds__(256)
void lc_loss(const float* __restrict__ z, const long long* __restrict__ neg_list,
             const int* __restrict__ top_idx, float* __restrict__ row_loss) {
  const int wave = threadIdx.x >> 5;
  const int lane = threadIdx.x & 31;
  const int row  = blockIdx.x * 8 + wave;

  v4f zi = *(const v4f*)(z + (size_t)row * DIMS + lane * 4);

  v4f acc = {};
  #pragma unroll
  for (int t = 0; t < TOPK; ++t) {
    int idx = top_idx[(size_t)row * TOPK + t];
    v4f p = *(const v4f*)(z + (size_t)idx * DIMS + lane * 4);
    acc += p;
  }
  int nidx = (int)neg_list[row];
  v4f nz = *(const v4f*)(z + (size_t)nidx * DIMS + lane * 4);

  float pd = 0.f, nd = 0.f;
  #pragma unroll
  for (int j = 0; j < 4; ++j) { pd += zi[j] * acc[j]; nd += zi[j] * nz[j]; }
  #pragma unroll
  for (int off = 16; off > 0; off >>= 1) {
    pd += __shfl_xor(pd, off);
    nd += __shfl_xor(nd, off);
  }

  if (lane == 0) {
    const float pl = pd * 0.2f;   // mean over 5 neighbors; mu = 1
    const float nl = nd;
    const float sp = 1.0f / (1.0f + expf(-pl));
    const float sn = 1.0f / (1.0f + expf(-nl));
    row_loss[row] = -logf(sp + 1e-15f) - logf(1.0f - sn + 1e-15f);
  }
}

// ---------------------------------------------------------------------------
// K4: deterministic reduction of per-row losses -> scalar mean-sum.
__global__ __launch_bounds__(256)
void lc_reduce(const float* __restrict__ row_loss, float* __restrict__ out) {
  __shared__ float sm[256];
  float s = 0.f;
  for (int i = threadIdx.x; i < N_ROWS; i += 256) s += row_loss[i];
  sm[threadIdx.x] = s;
  __syncthreads();
  #pragma unroll
  for (int off = 128; off > 0; off >>= 1) {
    if (threadIdx.x < off) sm[threadIdx.x] += sm[threadIdx.x + off];
    __syncthreads();
  }
  if (threadIdx.x == 0) out[0] = sm[0] / (float)N_ROWS;
}

// ---------------------------------------------------------------------------
extern "C" void kernel_launch(void* const* d_in, const int* in_sizes, int n_in,
                              void* d_out, int out_size, void* d_ws, size_t ws_size,
                              hipStream_t stream) {
  (void)in_sizes; (void)n_in; (void)out_size; (void)ws_size;
  const float*     z   = (const float*)d_in[0];
  const long long* neg = (const long long*)d_in[1];   // int64 in the reference

  char* ws = (char*)d_ws;
  _Float16* x    = (_Float16*)ws;                                   // 4 MB
  int*      topi = (int*)(ws + (size_t)N_ROWS * DIMS * 2);          // 320 KB
  float*    rls  = (float*)(ws + (size_t)N_ROWS * DIMS * 2
                               + (size_t)N_ROWS * TOPK * 4);        // 64 KB
  float*    out  = (float*)d_out;

  lc_normalize<<<N_ROWS / 8,  256, 0, stream>>>(z, x);
  lc_sim_topk <<<N_ROWS / 32,  32, 0, stream>>>(x, topi);
  lc_loss     <<<N_ROWS / 8,  256, 0, stream>>>(z, neg, topi, rls);
  lc_reduce   <<<1,           256, 0, stream>>>(rls, out);
}
